// EntityEmbbederKB_24790551232626
// MI455X (gfx1250) — matrix operand order repaired and checked by reading, not compiled, not run on hardware
//
#include <hip/hip_runtime.h>
#include <hip/hip_bf16.h>

typedef __attribute__((ext_vector_type(16))) __bf16 v16bf;
typedef __attribute__((ext_vector_type(8)))  float  v8f;

namespace {

constexpr int U_TOTAL  = 8192;   // unique entities
constexpr int FACTS    = 50;     // facts per entity
constexpr int MROWS    = 64;     // facts padded to 4 M-tiles of 16
constexpr int DIM      = 128;    // K = DP + DE
constexpr int HID      = 128;    // N = hidden
constexpr int EPB      = 16;     // entities per block (amortizes W staging)
constexpr int THREADS  = 256;    // 8 waves (wave32)
constexpr int ELEMS    = (MROWS * DIM) / THREADS;   // 32 staged elems/thread
constexpr int A_STRIDE = DIM + 8;   // +16B pad: frag b128 loads hit distinct banks
constexpr int B_STRIDE = 32 + 8;    // per-(kt,n) k-run, +16B pad

union Frag16 { uint4 q[2]; v16bf v; };

__global__ __launch_bounds__(THREADS)
void fact_mlp_pool(const float* __restrict__ p_table,
                   const float* __restrict__ e_table,
                   const float* __restrict__ W,
                   const float* __restrict__ bias,
                   const int* __restrict__ p_idx,
                   const int* __restrict__ e_idx,
                   float* __restrict__ pooled)
{
  __shared__ __bf16 sA[2][MROWS * A_STRIDE];     // double-buffered fact tiles
  __shared__ __bf16 sB[4 * HID * B_STRIDE];      // W in B-operand layout

  const int t    = threadIdx.x;
  const int wave = t >> 5;
  const int lane = t & 31;
  const int half = lane >> 4;
  const int l16  = lane & 15;

  // ---- Stage W (f32 -> bf16) into B-operand layout, once per block.
  // sB[(kt*HID + n)*B_STRIDE + k] = W[(kt*32+k)*HID + n]
  for (int i = 0; i < (DIM * HID) / THREADS; ++i) {
    int idx = i * THREADS + t;               // == d*HID + n (coalesced read of W)
    int d = idx >> 7, n = idx & (HID - 1);
    int kt = d >> 5, k = d & 31;
    sB[(kt * HID + n) * B_STRIDE + k] = (__bf16)W[idx];
  }
  __syncthreads();

  // ---- Preload this wave's B fragments (columns n = wave*16 + l16).
  // B 32x16, this code's contiguous k order (same as A): lanes 0-15 hold K=0..15, lanes 16-31 K=16..31.
  Frag16 bfrag[4];
  {
    const int n = (wave << 4) | l16;
#pragma unroll
    for (int kt = 0; kt < 4; ++kt) {
      const __bf16* bp = &sB[(kt * HID + n) * B_STRIDE + (half << 4)];
      bfrag[kt].q[0] = *(const uint4*)(bp);
      bfrag[kt].q[1] = *(const uint4*)(bp + 8);
    }
  }
  const float bval = bias[(wave << 4) | l16];
  const int   base = blockIdx.x * EPB;

  // ---- Prologue: stage entity `base` into buffer 0.
  for (int i = 0; i < ELEMS; ++i) {
    int idx = i * THREADS + t;
    int row = idx >> 7, col = idx & (DIM - 1);
    float v = 0.0f;
    if (row < FACTS) {
      if (col < 64) v = p_table[p_idx[base * FACTS + row] * 64 + col];
      else          v = e_table[e_idx[base * FACTS + row] * 64 + (col - 64)];
    }
    sA[0][row * A_STRIDE + col] = (__bf16)v;
  }
  __syncthreads();

  for (int e = 0; e < EPB; ++e) {
    const int cur = e & 1, nxt = cur ^ 1;
    const int u   = base + e;
    const bool haveNext = (e + 1 < EPB);

    // ---- Phase A: issue next entity's gather loads into registers so the
    // random-row global traffic is in flight underneath this entity's WMMAs.
    float vals[ELEMS];
    if (haveNext) {
      const int un = u + 1;
#pragma unroll
      for (int i = 0; i < ELEMS; ++i) {
        int idx = i * THREADS + t;
        int row = idx >> 7, col = idx & (DIM - 1);
        float v = 0.0f;
        if (row < FACTS) {
          if (col < 64) v = p_table[p_idx[un * FACTS + row] * 64 + col];
          else          v = e_table[e_idx[un * FACTS + row] * 64 + (col - 64)];
        }
        vals[i] = v;
      }
      // Prefetch entity e+2's rows toward L2 (one hint per 128B half-row).
      if (e + 2 < EPB) {
        const int un2  = u + 2;
        const int f    = t >> 2;
        const int sel  = (t >> 1) & 1;
        const int part = t & 1;
        if (f < FACTS) {
          const float* p = sel
              ? (e_table + (size_t)e_idx[un2 * FACTS + f] * 64)
              : (p_table + (size_t)p_idx[un2 * FACTS + f] * 64);
          __builtin_prefetch(p + part * 32, 0, 1);
        }
      }
    }

    // ---- Phase B: wave owns a 16-wide N-tile; per M-tile load ALL K-step
    // fragments first, then fire the 4 WMMAs back-to-back (one dscnt wait).
    float colmax = -3.4e38f;
#pragma unroll
    for (int mt = 0; mt < 4; ++mt) {
      const __bf16* arow = &sA[cur][((mt << 4) | l16) * A_STRIDE];
      Frag16 afrag[4];
#pragma unroll
      for (int kt = 0; kt < 4; ++kt) {
        const __bf16* ap = arow + (kt << 5) + (half << 4);
        afrag[kt].q[0] = *(const uint4*)(ap);      // K = kt*32 + half*16 + 0..7
        afrag[kt].q[1] = *(const uint4*)(ap + 8);  // K = kt*32 + half*16 + 8..15
      }
      v8f acc = {};
#pragma unroll
      for (int kt = 0; kt < 4; ++kt)
        acc = __builtin_amdgcn_wmma_f32_16x16x32_bf16(
            false, afrag[kt].v, false, bfrag[kt].v, (short)0, acc, false, false);
      // C/D layout: lane column = l16, rows = mt*16 + half*8 + r
      const int fbase = (mt << 4) + (half << 3);
#pragma unroll
      for (int r = 0; r < 8; ++r)
        if (fbase + r < FACTS) colmax = fmaxf(colmax, acc[r]);
    }
    // rows 0-7 live in lanes 0-15, rows 8-15 in lanes 16-31 -> merge halves.
    const float m = fmaxf(colmax, __shfl_xor(colmax, 16, 32));
    // max over facts commutes with monotone relu(x + b): apply once.
    const float result = fmaxf(m + bval, 0.0f);
    if (lane < 16)
      pooled[u * HID + (wave << 4) + l16] = result;

    // ---- Phase C: convert + commit staged registers into the other buffer.
    if (haveNext) {
#pragma unroll
      for (int i = 0; i < ELEMS; ++i) {
        int idx = i * THREADS + t;
        int row = idx >> 7, col = idx & (DIM - 1);
        sA[nxt][row * A_STRIDE + col] = (__bf16)vals[i];
      }
    }
    __syncthreads();   // one barrier per entity; buffers alternate safely
  }
}

__global__ __launch_bounds__(256)
void gather_rows(const float* __restrict__ pooled,
                 const int* __restrict__ cand,
                 float* __restrict__ out, int nrows)
{
  const int rowsPerBlock = blockDim.x >> 5;
  const int row = blockIdx.x * rowsPerBlock + (threadIdx.x >> 5);
  if (row >= nrows) return;
  const int lane = threadIdx.x & 31;
  const float4* src = (const float4*)(pooled + (size_t)cand[row] * HID);
  float4*       dst = (float4*)(out + (size_t)row * HID);
  dst[lane] = src[lane];   // 32 lanes x 16B = one 128-float row
}

} // anonymous namespace

extern "C" void kernel_launch(void* const* d_in, const int* in_sizes, int n_in,
                              void* d_out, int out_size, void* d_ws, size_t ws_size,
                              hipStream_t stream)
{
  const float* p_table = (const float*)d_in[0];
  const float* e_table = (const float*)d_in[1];
  const float* W       = (const float*)d_in[2];
  const float* bias    = (const float*)d_in[3];
  const int*   p_idx   = (const int*)d_in[4];
  const int*   e_idx   = (const int*)d_in[5];
  const int*   cand    = (const int*)d_in[6];
  float* out = (float*)d_out;

  float* pooled = (float*)d_ws;   // U_TOTAL * HID floats = 4 MB scratch

  fact_mlp_pool<<<U_TOTAL / EPB, THREADS, 0, stream>>>(
      p_table, e_table, W, bias, p_idx, e_idx, pooled);

  const int nrows = in_sizes[6];              // B*C = 2048*17 = 34816
  const int rowsPerBlock = 256 / 32;
  gather_rows<<<(nrows + rowsPerBlock - 1) / rowsPerBlock, 256, 0, stream>>>(
      pooled, cand, out, nrows);
}
